// MambaBlock_52905407152465
// MI455X (gfx1250) — matrix operand order repaired and checked by reading, not compiled, not run on hardware
//
#include <hip/hip_runtime.h>

#define D_MODEL  1024
#define D_FF     4096
#define D_STATE  16
#define D_CONV   4
#define D_INNER  2048
#define DT_RANK  64
#define SEQ_B    2
#define SEQ_L    4096
#define MTOT     (SEQ_B * SEQ_L)      // 8192 rows for every GEMM
#define DBC_W    (DT_RANK + 2 * D_STATE)   // 96

// ---------------- CDNA5 WMMA types ----------------
typedef __attribute__((ext_vector_type(16))) __bf16 v16bf;
typedef __attribute__((ext_vector_type(8)))  __bf16 v8bf;
typedef __attribute__((ext_vector_type(8)))  float  v8f;

union FragAB { v16bf v; v8bf h2[2]; __bf16 e[16]; };
union FragC  { v8f   v; float  e[8];  };

__device__ __forceinline__ float silu_f(float x) { return x / (1.f + __expf(-x)); }
__device__ __forceinline__ float softplus_f(float x) {
    return (x > 20.f) ? x : log1pf(__expf(x));
}

// =====================================================================
// Tiled bf16 WMMA GEMM:  C[M,N] = A[M,K] @ B[K,N], all dims compile-time.
// fp32 in HBM -> bf16 in LDS, f32 accumulate via V_WMMA_F32_16X16X32_BF16.
// Block = 256 threads = 8 waves; tile 128x128x32; wave = 4x2 16x16 tiles.
// LDS layouts make every fragment exactly two 16-byte ds_load_b128:
//   As[row][k]  (pad 40): A-frag lane(h,m) reads k = h*8..+7 and 16+h*8..+7
//   Bt[col][k]  (pad 40): B-frag lane(h,n) reads k = h*8..+7 and 16+h*8..+7
// Compile-time LDC makes epilogue stores base + immediate-IOFFSET.
// EPI: 0 = store, 1 = +bias then softplus, 2 = +residual
// NGUARD: column bounds checks (only the N=96 GEMM needs them)
// =====================================================================
#define BM 128
#define BN 128
#define BK 32
#define LDP (BK + 8)   // padded stride: 40 bf16 = 80B (16B-aligned, bank-spread)

template <int EPI, bool NGUARD, int LDA, int LDB, int LDC, int NN, int KK>
__global__ __launch_bounds__(256) void gemm_bf16_wmma(
    const float* __restrict__ A, const float* __restrict__ Bw,
    float* __restrict__ C,
    const float* __restrict__ extra)   // bias (EPI 1) or residual (EPI 2)
{
    __shared__ __align__(16) __bf16 As[BM][LDP];
    __shared__ __align__(16) __bf16 Bt[BN][LDP];   // transposed B tile

    const int tid  = threadIdx.x;
    const int wave = tid >> 5;
    const int lane = tid & 31;
    const int half = lane >> 4;         // wave32 lane-half per ISA layout
    const int lm   = lane & 15;
    const int wm   = (wave >> 2) * 64;  // wave row offset in block tile
    const int wn   = (wave & 3) * 32;   // wave col offset in block tile

    const int bm = blockIdx.y * BM;
    const int bn = blockIdx.x * BN;

    FragC acc[4][2];
#pragma unroll
    for (int mi = 0; mi < 4; ++mi)
#pragma unroll
        for (int ni = 0; ni < 2; ++ni)
#pragma unroll
            for (int r = 0; r < 8; ++r) acc[mi][ni].e[r] = 0.f;

    for (int kb = 0; kb < KK; kb += BK) {
        // ---- stage A tile 128x32 (4 float4 per thread), fp32 -> bf16 ----
#pragma unroll
        for (int i = 0; i < 4; ++i) {
            int idx = tid + i * 256;
            int r   = idx >> 3;
            int c4  = (idx & 7) * 4;
            float4 f = *(const float4*)(A + (size_t)(bm + r) * LDA + kb + c4);
            As[r][c4 + 0] = (__bf16)f.x;
            As[r][c4 + 1] = (__bf16)f.y;
            As[r][c4 + 2] = (__bf16)f.z;
            As[r][c4 + 3] = (__bf16)f.w;
        }
        // ---- stage B tile 32x128 transposed into Bt[col][k] ----
#pragma unroll
        for (int i = 0; i < 4; ++i) {
            int idx = tid + i * 256;
            int r   = idx >> 5;            // k within tile
            int c4  = (idx & 31) * 4;      // col within tile
            int col = bn + c4;
            const float* src = Bw + (size_t)(kb + r) * LDB + col;
            if (!NGUARD || (col + 4 <= NN)) {
                float4 f = *(const float4*)src;
                Bt[c4 + 0][r] = (__bf16)f.x;
                Bt[c4 + 1][r] = (__bf16)f.y;
                Bt[c4 + 2][r] = (__bf16)f.z;
                Bt[c4 + 3][r] = (__bf16)f.w;
            } else {
#pragma unroll
                for (int j = 0; j < 4; ++j)
                    Bt[c4 + j][r] = (col + j < NN) ? (__bf16)src[j] : (__bf16)0.f;
            }
        }
        __syncthreads();

        // ---- fragments: two ds_load_b128 each ----
        FragAB af[4], bf[2];
#pragma unroll
        for (int mi = 0; mi < 4; ++mi) {
            const __bf16* rp = &As[wm + mi * 16 + lm][0];
            af[mi].h2[0] = *(const v8bf*)(rp + half * 8);        // k = h*8..+7
            af[mi].h2[1] = *(const v8bf*)(rp + 16 + half * 8);   // k = 16+h*8..+7
        }
#pragma unroll
        for (int ni = 0; ni < 2; ++ni) {
            const __bf16* cp = &Bt[wn + ni * 16 + lm][0];
            bf[ni].h2[0] = *(const v8bf*)(cp + half * 8);        // k = h*8..+7
            bf[ni].h2[1] = *(const v8bf*)(cp + 16 + half * 8);   // k = 16+h*8..+7
        }

        // ---- 8 WMMAs: D = A x B + C ----
#pragma unroll
        for (int mi = 0; mi < 4; ++mi)
#pragma unroll
            for (int ni = 0; ni < 2; ++ni)
                acc[mi][ni].v = __builtin_amdgcn_wmma_f32_16x16x32_bf16(
                    false, af[mi].v, false, bf[ni].v,
                    (short)0, acc[mi][ni].v, false, false);
        __syncthreads();
    }

    // ---- epilogue + store (C frag: VGPR r -> row r + half*8, col = lane%16) ----
    // LDC is compile-time: r*LDC folds into the store's immediate offset.
#pragma unroll
    for (int mi = 0; mi < 4; ++mi)
#pragma unroll
        for (int ni = 0; ni < 2; ++ni) {
            const int col  = bn + wn + ni * 16 + lm;
            if (NGUARD && col >= NN) continue;
            const int row0 = bm + wm + mi * 16 + half * 8;
            float* cp = C + (size_t)row0 * LDC + col;
            const float* rp = (EPI == 2) ? (extra + (size_t)row0 * LDC + col)
                                         : nullptr;
#pragma unroll
            for (int r = 0; r < 8; ++r) {
                float v = acc[mi][ni].e[r];
                if (EPI == 1)      v = softplus_f(v + extra[col]);
                else if (EPI == 2) v += rp[r * LDC];
                cp[r * LDC] = v;
            }
        }
}

// =====================================================================
// RMSNorm: one block per row of 1024, 256 threads x float4
// =====================================================================
__global__ __launch_bounds__(256) void rmsnorm_kernel(
    const float* __restrict__ x, const float* __restrict__ w,
    float* __restrict__ out)
{
    const size_t row = blockIdx.x;
    const float4 v = ((const float4*)(x + row * D_MODEL))[threadIdx.x];
    float ss = v.x * v.x + v.y * v.y + v.z * v.z + v.w * v.w;
#pragma unroll
    for (int off = 16; off > 0; off >>= 1) ss += __shfl_down(ss, off, 32);
    __shared__ float wsum[8];
    if ((threadIdx.x & 31) == 0) wsum[threadIdx.x >> 5] = ss;
    __syncthreads();
    float tot = wsum[0] + wsum[1] + wsum[2] + wsum[3] +
                wsum[4] + wsum[5] + wsum[6] + wsum[7];
    float scale = rsqrtf(tot * (1.f / D_MODEL) + 1e-6f);
    const float4 wv = ((const float4*)w)[threadIdx.x];
    float4 o;
    o.x = v.x * scale * wv.x; o.y = v.y * scale * wv.y;
    o.z = v.z * scale * wv.z; o.w = v.w * scale * wv.w;
    ((float4*)(out + row * D_MODEL))[threadIdx.x] = o;
}

// =====================================================================
// Depthwise causal conv1d (k=4) + SiLU over u = xz[:, :, :D_INNER]
// =====================================================================
__global__ __launch_bounds__(256) void conv_silu_kernel(
    const float* __restrict__ xz, const float* __restrict__ cw,
    const float* __restrict__ cb, float* __restrict__ uc)
{
    const size_t gid = (size_t)blockIdx.x * 256 + threadIdx.x;
    const int c  = gid & (D_INNER - 1);
    const size_t bl = gid >> 11;           // b*L + l
    const int l  = (int)(bl & (SEQ_L - 1));
    float acc = cb[c];
#pragma unroll
    for (int k = 0; k < D_CONV; ++k) {
        int lk = l + k - (D_CONV - 1);
        if (lk >= 0)
            acc += xz[(bl + (size_t)(lk - l)) * (2 * D_INNER) + c] * cw[c * D_CONV + k];
    }
    uc[gid] = silu_f(acc);
}

// =====================================================================
// Selective scan: thread per (b, d) channel, d_state=16 in registers.
// Fuses  y = (scan + u*D) * silu(z)
// =====================================================================
__global__ __launch_bounds__(256) void scan_kernel(
    const float* __restrict__ delta, const float* __restrict__ dbc,
    const float* __restrict__ uc, const float* __restrict__ xz,
    const float* __restrict__ A_log, const float* __restrict__ Dw,
    float* __restrict__ y)
{
    const int t = blockIdx.x * 256 + threadIdx.x;   // 0 .. B*D_INNER-1
    const int d = t & (D_INNER - 1);
    const int b = t >> 11;
    float Aa[D_STATE], h[D_STATE];
#pragma unroll
    for (int s = 0; s < D_STATE; ++s) {
        Aa[s] = -__expf(A_log[d * D_STATE + s]);
        h[s]  = 0.f;
    }
    const float Dv = Dw[d];
    for (int l = 0; l < SEQ_L; ++l) {
        const size_t bl = (size_t)b * SEQ_L + l;
        const float dt = delta[bl * D_INNER + d];
        const float u  = uc[bl * D_INNER + d];
        const float* bc = dbc + bl * DBC_W;
        const float du = dt * u;
        float yv = 0.f;
#pragma unroll
        for (int s = 0; s < D_STATE; ++s) {
            h[s] = h[s] * __expf(dt * Aa[s]) + du * bc[DT_RANK + s];
            yv += h[s] * bc[DT_RANK + D_STATE + s];
        }
        const float z = xz[bl * (2 * D_INNER) + D_INNER + d];
        y[bl * D_INNER + d] = (yv + u * Dv) * silu_f(z);
    }
}

// =====================================================================
// SwiGLU gate (in place): g1 = silu(g1) * g3
// =====================================================================
__global__ __launch_bounds__(256) void swiglu_kernel(
    float* __restrict__ g1, const float* __restrict__ g3)
{
    const size_t i = (size_t)blockIdx.x * 256 + threadIdx.x;
    g1[i] = silu_f(g1[i]) * g3[i];
}

// =====================================================================
// Host launcher
// =====================================================================
extern "C" void kernel_launch(void* const* d_in, const int* in_sizes, int n_in,
                              void* d_out, int out_size, void* d_ws, size_t ws_size,
                              hipStream_t stream)
{
    (void)in_sizes; (void)n_in; (void)out_size; (void)ws_size;
    const float* x         = (const float*)d_in[0];
    const float* n1_w      = (const float*)d_in[1];
    const float* in_proj_w = (const float*)d_in[2];
    const float* conv_w    = (const float*)d_in[3];
    const float* conv_b    = (const float*)d_in[4];
    const float* x_proj_w  = (const float*)d_in[5];
    const float* dt_proj_w = (const float*)d_in[6];
    const float* dt_proj_b = (const float*)d_in[7];
    const float* A_log     = (const float*)d_in[8];
    const float* Dv        = (const float*)d_in[9];
    const float* out_proj_w= (const float*)d_in[10];
    const float* n2_w      = (const float*)d_in[11];
    const float* w1        = (const float*)d_in[12];
    const float* w2        = (const float*)d_in[13];
    const float* w3        = (const float*)d_in[14];
    float* out = (float*)d_out;

    // workspace layout (floats), with lifetime-based reuse
    float* ws = (float*)d_ws;
    const size_t SZ_XZ  = (size_t)MTOT * 2 * D_INNER;   // 33.5M
    const size_t SZ_XN  = (size_t)MTOT * D_MODEL;       //  8.4M
    const size_t SZ_IN  = (size_t)MTOT * D_INNER;       // 16.8M
    float* xz    = ws;                        // later reused as g1
    float* xn    = xz  + SZ_XZ;               // later reused as h
    float* uc    = xn  + SZ_XN;               // uc + delta later reused as g3
    float* delta = uc  + SZ_IN;
    float* ybuf  = delta + SZ_IN;
    float* dbc   = ybuf + SZ_IN;
    float* g1 = xz;
    float* g3 = uc;      // spans uc(16.8M)+delta(16.8M) = 33.5M floats
    float* h  = xn;

    const dim3 blk(256);

    // 1) xn = rmsnorm(x, n1_w)
    rmsnorm_kernel<<<MTOT, blk, 0, stream>>>(x, n1_w, xn);
    // 2) xz = xn @ in_proj_w            [8192 x 4096], K=1024
    gemm_bf16_wmma<0, false, D_MODEL, 2 * D_INNER, 2 * D_INNER, 2 * D_INNER, D_MODEL>
        <<<dim3(2 * D_INNER / BN, MTOT / BM), blk, 0, stream>>>(
            xn, in_proj_w, xz, nullptr);
    // 3) uc = silu(causal_conv(u))
    conv_silu_kernel<<<(unsigned)((size_t)MTOT * D_INNER / 256), blk, 0, stream>>>(
        xz, conv_w, conv_b, uc);
    // 4) dbc = uc @ x_proj_w            [8192 x 96], K=2048  (guarded N)
    gemm_bf16_wmma<0, true, D_INNER, DBC_W, DBC_W, DBC_W, D_INNER>
        <<<dim3(1, MTOT / BM), blk, 0, stream>>>(
            uc, x_proj_w, dbc, nullptr);
    // 5) delta = softplus(dt @ dt_proj_w + b)   (dt = dbc[:, :64], lda=96)
    gemm_bf16_wmma<1, false, DBC_W, D_INNER, D_INNER, D_INNER, DT_RANK>
        <<<dim3(D_INNER / BN, MTOT / BM), blk, 0, stream>>>(
            dbc, dt_proj_w, delta, dt_proj_b);
    // 6) ybuf = (scan(...) + u*D) * silu(z)
    scan_kernel<<<SEQ_B * D_INNER / 256, blk, 0, stream>>>(
        delta, dbc, uc, xz, A_log, Dv, ybuf);
    // 7) out = x + ybuf @ out_proj_w    [8192 x 1024], K=2048  (residual)
    gemm_bf16_wmma<2, false, D_INNER, D_MODEL, D_MODEL, D_MODEL, D_INNER>
        <<<dim3(D_MODEL / BN, MTOT / BM), blk, 0, stream>>>(
            ybuf, out_proj_w, out, x);
    // 8) h = rmsnorm(out, n2_w)
    rmsnorm_kernel<<<MTOT, blk, 0, stream>>>(out, n2_w, h);
    // 9) g1 = h @ w1 ; 10) g3 = h @ w3   [8192 x 4096], K=1024 (same shape as in_proj)
    gemm_bf16_wmma<0, false, D_MODEL, D_FF, D_FF, D_FF, D_MODEL>
        <<<dim3(D_FF / BN, MTOT / BM), blk, 0, stream>>>(
            h, w1, g1, nullptr);
    gemm_bf16_wmma<0, false, D_MODEL, D_FF, D_FF, D_FF, D_MODEL>
        <<<dim3(D_FF / BN, MTOT / BM), blk, 0, stream>>>(
            h, w3, g3, nullptr);
    // 11) g1 = silu(g1) * g3
    swiglu_kernel<<<(unsigned)((size_t)MTOT * D_FF / 256), blk, 0, stream>>>(g1, g3);
    // 12) out = out + g1 @ w2           [8192 x 1024], K=4096  (residual, aliased safely)
    gemm_bf16_wmma<2, false, D_FF, D_MODEL, D_MODEL, D_MODEL, D_FF>
        <<<dim3(D_MODEL / BN, MTOT / BM), blk, 0, stream>>>(
            g1, w2, out, out);
}
